// SimpleMoEBlock_49082886258849
// MI455X (gfx1250) — matrix operand order repaired and checked by reading, not compiled, not run on hardware
//
#include <hip/hip_runtime.h>
#include <hip/hip_bf16.h>

// ---------- problem constants ----------
constexpr int T_TOK = 4096;   // tokens
constexpr int HID   = 1024;   // hidden
constexpr int INTER = 4096;   // intermediate
constexpr int NEXP  = 8;      // experts

// ---------- vector types ----------
typedef __attribute__((ext_vector_type(16))) __bf16        v16bf;
typedef __attribute__((ext_vector_type(8)))  float         v8f;
typedef __attribute__((ext_vector_type(4)))  unsigned int  u32x4;
typedef __attribute__((ext_vector_type(2)))  unsigned int  u32x2;

// ---------- GEMM tiling ----------
constexpr int BK = 32;
constexpr int PITCH = 40;   // bf16 elements per LDS row (80 bytes, 16B aligned, bank-spread pad)

// =====================================================================
// Gate: logits = x @ Wg^T  (per token, 8 experts), softmax over experts
// =====================================================================
__global__ __launch_bounds__(256)
void gate_softmax(const float* __restrict__ x, const float* __restrict__ Wg,
                  float* __restrict__ gate)
{
    __shared__ float lg[256];
    const int tid = threadIdx.x;
    const int t = blockIdx.x * 32 + (tid >> 3);
    const int e = tid & 7;
    const float* xr = x  + (size_t)t * HID;
    const float* wr = Wg + (size_t)e * HID;
    float acc = 0.f;
    for (int h = 0; h < HID; h += 4) {
        float4 a = *(const float4*)(xr + h);
        float4 w = *(const float4*)(wr + h);
        acc += a.x * w.x + a.y * w.y + a.z * w.z + a.w * w.w;
    }
    lg[tid] = acc;
    __syncthreads();
    const int base = tid & ~7;
    float mx = lg[base];
#pragma unroll
    for (int j = 1; j < 8; j++) mx = fmaxf(mx, lg[base + j]);
    float s = 0.f;
#pragma unroll
    for (int j = 0; j < 8; j++) s += __expf(lg[base + j] - mx);
    gate[(size_t)t * NEXP + e] = __expf(acc - mx) / s;
}

// =====================================================================
// Elementwise f32 -> bf16 (8 elems / thread)
// =====================================================================
__global__ __launch_bounds__(256)
void cvt_f32_bf16(const float* __restrict__ in, __bf16* __restrict__ out)
{
    const size_t i = ((size_t)blockIdx.x * 256 + threadIdx.x) * 8;
    float4 a = *(const float4*)(in + i);
    float4 b = *(const float4*)(in + i + 4);
    union { __bf16 h[8]; u32x4 u; } pk;
    pk.h[0] = (__bf16)a.x; pk.h[1] = (__bf16)a.y; pk.h[2] = (__bf16)a.z; pk.h[3] = (__bf16)a.w;
    pk.h[4] = (__bf16)b.x; pk.h[5] = (__bf16)b.y; pk.h[6] = (__bf16)b.z; pk.h[7] = (__bf16)b.w;
    *(u32x4*)(out + i) = pk.u;
}

// =====================================================================
// Transpose + convert: W[e][R][C] f32  ->  WT[e][C][R] bf16  (64x64 LDS tiles)
// =====================================================================
__global__ __launch_bounds__(256)
void transpose_cvt(const float* __restrict__ W, __bf16* __restrict__ WT, int R, int C)
{
    __shared__ float tile[64][65];
    const int e  = blockIdx.z;
    const float* src = W  + (size_t)e * R * C;
    __bf16*      dst = WT + (size_t)e * R * C;
    const int r0 = blockIdx.y * 64, c0 = blockIdx.x * 64;
    const int tid = threadIdx.x;
    const int tr = tid >> 4;          // 0..15
    const int tc = (tid & 15) << 2;   // 0..60 step 4
#pragma unroll
    for (int i = 0; i < 4; i++) {
        const int rr = tr + i * 16;
        float4 v = *(const float4*)(src + (size_t)(r0 + rr) * C + c0 + tc);
        tile[rr][tc + 0] = v.x; tile[rr][tc + 1] = v.y;
        tile[rr][tc + 2] = v.z; tile[rr][tc + 3] = v.w;
    }
    __syncthreads();
#pragma unroll
    for (int i = 0; i < 4; i++) {
        const int cc = tr + i * 16;   // output row = original column
        union { __bf16 h[4]; u32x2 u; } pk;
#pragma unroll
        for (int j = 0; j < 4; j++) pk.h[j] = (__bf16)tile[tc + j][cc];
        *(u32x2*)(dst + (size_t)(c0 + cc) * R + r0 + tc) = pk.u;
    }
}

// =====================================================================
// WMMA GEMM:  C[M,N] = A[M,K](bf16, row-major) x B[N,K](bf16, n-major = pre-transposed)
// Double-buffered LDS pipeline, one barrier per K-step:
//   load_global(k+1) -> compute(buf k) -> store_lds(buf k+1) -> barrier
// Each wave owns a 64x64 C tile = 4x4 WMMA accumulators (16 WMMA : 16 ds_load_b128).
// EPI==0: out_bf16 = bf16( gelu(C + bias[n]) )          (GEMM1)
// EPI==1: out_f32 (+)= gate[m][e] * (C + bias[n])       (GEMM2, accumulate over experts)
// =====================================================================
template <int EPI, int BMt, int BNt, int WMG, int WNG>
__global__ __launch_bounds__(32 * WMG * WNG)
void gemm_wmma(const __bf16* __restrict__ A, const __bf16* __restrict__ Bw,
               int M, int N, int K,
               const float* __restrict__ bias,
               const float* __restrict__ gate, int e, int first,
               float* __restrict__ outF, __bf16* __restrict__ outH)
{
    constexpr int NT   = 32 * WMG * WNG;      // threads per block
    constexpr int FM   = BMt / WMG / 16;      // M fragments per wave (=4)
    constexpr int FN   = BNt / WNG / 16;      // N fragments per wave (=4)
    constexpr int SEGA = BMt * 2 / NT;        // 32B staging segments per thread (A)
    constexpr int SEGB = BNt * 2 / NT;        // 32B staging segments per thread (B)
    constexpr int ABUF = BMt * PITCH;         // LDS elements per A buffer
    constexpr int BBUF = BNt * PITCH;         // LDS elements per B buffer

    __shared__ __bf16 As[2 * ABUF];
    __shared__ __bf16 Bs[2 * BBUF];

    const int tid  = threadIdx.x;
    const int lane = tid & 31;
    const int wave = tid >> 5;
    const int wm = wave / WNG;
    const int wn = wave % WNG;
    const int half = lane >> 4;      // lane half (K-half select)
    const int lc   = lane & 15;      // M/N index within fragment
    const int m0 = blockIdx.y * BMt;
    const int n0 = blockIdx.x * BNt;

    v8f c[FM][FN] = {};

    // staging: 2 threads per 64B row, 32B (=16 bf16) per segment
    const __bf16* gA[SEGA];  int lA[SEGA];
    const __bf16* gB[SEGB];  int lB[SEGB];
#pragma unroll
    for (int s = 0; s < SEGA; s++) {
        const int sid = s * NT + tid, row = sid >> 1, seg = (sid & 1) << 4;
        gA[s] = A + (size_t)(m0 + row) * K + seg;
        lA[s] = row * PITCH + seg;
    }
#pragma unroll
    for (int s = 0; s < SEGB; s++) {
        const int sid = s * NT + tid, row = sid >> 1, seg = (sid & 1) << 4;
        gB[s] = Bw + (size_t)(n0 + row) * K + seg;
        lB[s] = row * PITCH + seg;
    }

    u32x4 ta[SEGA * 2], tb[SEGB * 2];

    auto load_global = [&](int it) {
        const int kk = it * BK;
#pragma unroll
        for (int s = 0; s < SEGA; s++) {
            ta[2 * s]     = *(const u32x4*)(gA[s] + kk);
            ta[2 * s + 1] = *(const u32x4*)(gA[s] + kk + 8);
        }
#pragma unroll
        for (int s = 0; s < SEGB; s++) {
            tb[2 * s]     = *(const u32x4*)(gB[s] + kk);
            tb[2 * s + 1] = *(const u32x4*)(gB[s] + kk + 8);
        }
    };
    auto store_lds = [&](int buf) {
        __bf16* ba = &As[buf * ABUF];
        __bf16* bb = &Bs[buf * BBUF];
#pragma unroll
        for (int s = 0; s < SEGA; s++) {
            *(u32x4*)(ba + lA[s])     = ta[2 * s];
            *(u32x4*)(ba + lA[s] + 8) = ta[2 * s + 1];
        }
#pragma unroll
        for (int s = 0; s < SEGB; s++) {
            *(u32x4*)(bb + lB[s])     = tb[2 * s];
            *(u32x4*)(bb + lB[s] + 8) = tb[2 * s + 1];
        }
    };
    auto compute = [&](int buf) {
        const __bf16* ba = &As[buf * ABUF];
        const __bf16* bb = &Bs[buf * BBUF];
        union Frag { u32x4 q[2]; v16bf v; } a[FM], b[FN];
        // A fragment: 16x32 bf16, lane lc = row M, K split per ISA swizzle:
        //   VGPR0-3 : K = half*8 + 0..7   -> bytes [half*16, +16)
        //   VGPR4-7 : K = 16+half*8 ..    -> bytes [32+half*16, +16)
#pragma unroll
        for (int mi = 0; mi < FM; mi++) {
            const __bf16* p = ba + (wm * (FM * 16) + mi * 16 + lc) * PITCH;
            a[mi].q[0] = *(const u32x4*)(p + half * 8);
            a[mi].q[1] = *(const u32x4*)(p + 16 + half * 8);
        }
        // B fragment: 32x16 bf16, lane lc = col N; K split same as A (half*8 + 0..7, 16+half*8 + 0..7)
#pragma unroll
        for (int ni = 0; ni < FN; ni++) {
            const __bf16* p = bb + (wn * (FN * 16) + ni * 16 + lc) * PITCH;
            b[ni].q[0] = *(const u32x4*)(p + half * 8);
            b[ni].q[1] = *(const u32x4*)(p + 16 + half * 8);
        }
#pragma unroll
        for (int mi = 0; mi < FM; mi++)
#pragma unroll
            for (int ni = 0; ni < FN; ni++)
                c[mi][ni] = __builtin_amdgcn_wmma_f32_16x16x32_bf16(
                    false, a[mi].v, false, b[ni].v,
                    (short)0, c[mi][ni], false, false);
    };

    // ---- pipelined main loop (double-buffered LDS, one barrier per K-step) ----
    const int NIT = K / BK;
    load_global(0);
    store_lds(0);
    __syncthreads();
    for (int it = 0; it < NIT - 1; it++) {
        load_global(it + 1);
        compute(it & 1);
        store_lds((it + 1) & 1);
        __syncthreads();
    }
    compute((NIT - 1) & 1);

    // Epilogue. C layout: VGPR r, lanes 0-15 -> M=r, lanes 16-31 -> M=8+r, N=lc
#pragma unroll
    for (int mi = 0; mi < FM; mi++)
#pragma unroll
        for (int ni = 0; ni < FN; ni++)
#pragma unroll
            for (int r = 0; r < 8; r++) {
                const int grow = m0 + wm * (FM * 16) + mi * 16 + half * 8 + r;
                const int gcol = n0 + wn * (FN * 16) + ni * 16 + lc;
                float v = c[mi][ni][r] + bias[gcol];
                if (EPI == 0) {
                    v = 0.5f * v * (1.0f + erff(v * 0.70710678118654752f));
                    outH[(size_t)grow * N + gcol] = (__bf16)v;
                } else {
                    const float g = gate[(size_t)grow * NEXP + e];
                    float* po = &outF[(size_t)grow * N + gcol];
                    v *= g;
                    if (first) *po = v;
                    else       *po += v;
                }
            }
}

// =====================================================================
// launcher
// =====================================================================
extern "C" void kernel_launch(void* const* d_in, const int* in_sizes, int n_in,
                              void* d_out, int out_size, void* d_ws, size_t ws_size,
                              hipStream_t stream)
{
    const float* x  = (const float*)d_in[0];
    const float* Wg = (const float*)d_in[1];
    const float* W1 = (const float*)d_in[2];
    const float* b1 = (const float*)d_in[3];
    const float* W2 = (const float*)d_in[4];
    const float* b2 = (const float*)d_in[5];
    float* out = (float*)d_out;

    char* ws = (char*)d_ws;
    size_t ofs = 0;
    float*  gate = (float*)(ws + ofs);  ofs += (size_t)T_TOK * NEXP * sizeof(float);
    __bf16* xb   = (__bf16*)(ws + ofs); ofs += (size_t)T_TOK * HID * sizeof(__bf16);
    __bf16* w1t  = (__bf16*)(ws + ofs); ofs += (size_t)NEXP * HID * INTER * sizeof(__bf16);
    __bf16* w2t  = (__bf16*)(ws + ofs); ofs += (size_t)NEXP * HID * INTER * sizeof(__bf16);
    __bf16* hb   = (__bf16*)(ws + ofs); ofs += (size_t)T_TOK * INTER * sizeof(__bf16);

    // 1) gate probabilities
    gate_softmax<<<T_TOK / 32, 256, 0, stream>>>(x, Wg, gate);
    // 2) x -> bf16
    cvt_f32_bf16<<<(T_TOK * HID) / (256 * 8), 256, 0, stream>>>(x, xb);
    // 3) W1[e][H][I] -> w1t[e][I][H] bf16 ; W2[e][I][H] -> w2t[e][H][I] bf16
    transpose_cvt<<<dim3(INTER / 64, HID / 64, NEXP), 256, 0, stream>>>(W1, w1t, HID, INTER);
    transpose_cvt<<<dim3(HID / 64, INTER / 64, NEXP), 256, 0, stream>>>(W2, w2t, INTER, HID);

    // 4) per-expert fused GEMM1(+gelu) then GEMM2(+gate accumulate)
    // GEMM1: M=4096,N=4096,K=1024 : 128x256 tiles, 8 waves (2x4) -> 512 WGs
    // GEMM2: M=4096,N=1024,K=4096 : 128x128 tiles, 4 waves (2x2) -> 256 WGs
    for (int e = 0; e < NEXP; e++) {
        gemm_wmma<0, 128, 256, 2, 4><<<dim3(INTER / 256, T_TOK / 128), 256, 0, stream>>>(
            xb, w1t + (size_t)e * HID * INTER, T_TOK, INTER, HID,
            b1 + (size_t)e * INTER, nullptr, e, 0, nullptr, hb);
        gemm_wmma<1, 128, 128, 2, 2><<<dim3(HID / 128, T_TOK / 128), 128, 0, stream>>>(
            hb, w2t + (size_t)e * HID * INTER, T_TOK, HID, INTER,
            b2 + (size_t)e * HID, gate, e, (e == 0) ? 1 : 0, out, nullptr);
    }
}
